// TransformerBlock_83391085019730
// MI455X (gfx1250) — matrix operand order repaired and checked by reading, not compiled, not run on hardware
//
#include <hip/hip_runtime.h>

// ---------------------------------------------------------------------------
// Types for CDNA5 WMMA (wave32)
// ---------------------------------------------------------------------------
typedef __bf16 bf16_t;
typedef __attribute__((ext_vector_type(16))) __bf16 v16bf;
typedef __attribute__((ext_vector_type(8)))  __bf16 bf16x8;
typedef __attribute__((ext_vector_type(8)))  float  v8f;

#define SHUF16(lo, hi) __builtin_shufflevector((lo), (hi), 0,1,2,3,4,5,6,7,8,9,10,11,12,13,14,15)
#define WMMA_BF16(a, b, c) \
    __builtin_amdgcn_wmma_f32_16x16x32_bf16(false, (a), false, (b), (short)0, (c), false, false)

static constexpr int Bc = 4, Sc = 2048, Dc = 1024, Hc = 16, HDc = 64, Fc = 4096;
static constexpr int Mrows = Bc * Sc; // 8192

__device__ __forceinline__ bf16_t f2bf(float f) {
    unsigned u = __builtin_bit_cast(unsigned, f);
    unsigned r = (u + 0x7fffu + ((u >> 16) & 1u)) >> 16;
    unsigned short s = (unsigned short)r;
    return __builtin_bit_cast(bf16_t, s);
}

__device__ __forceinline__ v8f zero_v8f() {
    v8f z;
#pragma unroll
    for (int i = 0; i < 8; ++i) z[i] = 0.f;
    return z;
}

// Async global->LDS B128 copy (CDNA5 TDM-adjacent async path, tracked by ASYNCcnt).
// ldsOff: byte offset within the workgroup LDS allocation (low 32 bits of a
// generic pointer to a __shared__ object). gaddr: 64-bit global address.
__device__ __forceinline__ void async_copy_b128(unsigned ldsOff, const void* gaddr) {
    asm volatile("global_load_async_to_lds_b128 %0, %1, off"
                 :: "v"(ldsOff), "v"((unsigned long long)(uintptr_t)gaddr)
                 : "memory");
}
__device__ __forceinline__ void wait_async0() {
    asm volatile("s_wait_asynccnt 0x0" ::: "memory");
}
__device__ __forceinline__ unsigned lds_off(const void* p) {
    return (unsigned)(uintptr_t)p; // generic LDS pointer: low dword = LDS offset
}

// ---------------------------------------------------------------------------
// f32 -> bf16 conversion (weights)
// ---------------------------------------------------------------------------
__global__ __launch_bounds__(256) void f32_to_bf16_kernel(const float* __restrict__ in,
                                                          bf16_t* __restrict__ out, int n) {
    int i = blockIdx.x * 256 + threadIdx.x;
    if (i < n) out[i] = f2bf(in[i]);
}

// ---------------------------------------------------------------------------
// LayerNorm over last dim (D=1024), gamma=1 beta=0, eps=1e-4. f32 in, bf16 out.
// ---------------------------------------------------------------------------
__global__ __launch_bounds__(256) void ln_kernel(const float* __restrict__ x,
                                                 bf16_t* __restrict__ y) {
    __shared__ float ssum[256];
    __shared__ float ssq[256];
    const int tid = threadIdx.x;
    const size_t row = blockIdx.x;
    const float4 xv = ((const float4*)(x + row * Dc))[tid];
    float s = xv.x + xv.y + xv.z + xv.w;
    float q = xv.x * xv.x + xv.y * xv.y + xv.z * xv.z + xv.w * xv.w;
    ssum[tid] = s; ssq[tid] = q;
    __syncthreads();
#pragma unroll
    for (int o = 128; o > 0; o >>= 1) {
        if (tid < o) { ssum[tid] += ssum[tid + o]; ssq[tid] += ssq[tid + o]; }
        __syncthreads();
    }
    const float mean = ssum[0] * (1.f / Dc);
    const float var  = ssq[0] * (1.f / Dc) - mean * mean;
    const float inv  = rsqrtf(var + 1e-4f);
    bf16_t* yr = y + row * Dc + tid * 4;
    yr[0] = f2bf((xv.x - mean) * inv);
    yr[1] = f2bf((xv.y - mean) * inv);
    yr[2] = f2bf((xv.z - mean) * inv);
    yr[3] = f2bf((xv.w - mean) * inv);
}

// ---------------------------------------------------------------------------
// V transpose: (B,S,D) bf16 -> (B,H,HD,S) bf16 so P*V B-operand is contiguous.
// ---------------------------------------------------------------------------
__global__ __launch_bounds__(256) void transpose_v_kernel(const bf16_t* __restrict__ v,
                                                          bf16_t* __restrict__ vt) {
    size_t i = (size_t)blockIdx.x * 256 + threadIdx.x; // over B*S*D
    int d = (int)(i & (Dc - 1));
    size_t bs = i >> 10;
    int s = (int)(bs & (Sc - 1));
    int b = (int)(bs >> 11);
    int h = d >> 6, hd = d & 63;
    vt[(((size_t)b * Hc + h) * HDc + hd) * Sc + s] = v[i];
}

// ---------------------------------------------------------------------------
// NT GEMM: C[M,N] = A[M,K] * Bw[N,K]^T with WMMA bf16 16x16x32.
// Block tile 128x128, BK=32, 256 threads (8 waves, 2x4), wave tile 64x32.
// Tiles staged global->LDS via GLOBAL_LOAD_ASYNC_TO_LDS_B128 (ASYNCcnt),
// double-buffered: tile k+1 DMA overlaps tile k WMMAs; one barrier/iter.
// EPI: 0 = store bf16
//      1 = +bias, relu, store bf16
//      2 = +bias, +residual(f32), store f32
// ---------------------------------------------------------------------------
template <int EPI>
__global__ __launch_bounds__(256) void gemm_nt(const bf16_t* __restrict__ A,
                                               const bf16_t* __restrict__ Bw,
                                               const float* __restrict__ bias,
                                               const float* __restrict__ res,
                                               bf16_t* __restrict__ outB,
                                               float* __restrict__ outF,
                                               int M, int N, int K) {
    __shared__ __align__(16) bf16_t As[2][128][40]; // padded rows: 80B stride
    __shared__ __align__(16) bf16_t Bs[2][128][40];

    const int tid  = threadIdx.x;
    const int lane = tid & 31;
    const int wid  = tid >> 5;
    const int waveM = wid >> 2; // 0..1
    const int waveN = wid & 3;  // 0..3
    const int g    = lane >> 4;
    const int ln16 = lane & 15;
    const int mBase = blockIdx.y * 128;
    const int nBase = blockIdx.x * 128;

    v8f acc[4][2];
#pragma unroll
    for (int ms = 0; ms < 4; ++ms)
#pragma unroll
        for (int ns = 0; ns < 2; ++ns) acc[ms][ns] = zero_v8f();

    // each thread async-copies 2x b128 for A and for B per tile
    const int r0  = tid >> 2;        // 0..63
    const int q0i = (tid & 3) * 8;   // element col within 32-wide K tile

    auto gasync = [&](int kt, int buf) {
        const bf16_t* ga = A + (size_t)(mBase + r0) * K + kt * 32 + q0i;
        const bf16_t* gb = Bw + (size_t)(nBase + r0) * K + kt * 32 + q0i;
        async_copy_b128(lds_off(&As[buf][r0][q0i]),      ga);
        async_copy_b128(lds_off(&As[buf][r0 + 64][q0i]), ga + (size_t)64 * K);
        async_copy_b128(lds_off(&Bs[buf][r0][q0i]),      gb);
        async_copy_b128(lds_off(&Bs[buf][r0 + 64][q0i]), gb + (size_t)64 * K);
    };

    const int steps = K >> 5;
    gasync(0, 0);
    wait_async0();
    __syncthreads();

    for (int kt = 0; kt < steps; ++kt) {
        const int buf = kt & 1;
        // kick off DMA for next tile; overlaps with this tile's WMMAs
        if (kt + 1 < steps) gasync(kt + 1, buf ^ 1);

        v16bf af[4];
#pragma unroll
        for (int ms = 0; ms < 4; ++ms) {
            const int mr = waveM * 64 + ms * 16 + ln16;
            bf16x8 lo = *(const bf16x8*)&As[buf][mr][8 * g];
            bf16x8 hi = *(const bf16x8*)&As[buf][mr][8 * g + 16];
            af[ms] = SHUF16(lo, hi);
        }
        v16bf bfg[2];
#pragma unroll
        for (int ns = 0; ns < 2; ++ns) {
            const int nr = waveN * 32 + ns * 16 + ln16;
            bf16x8 lo = *(const bf16x8*)&Bs[buf][nr][8 * g];
            bf16x8 hi = *(const bf16x8*)&Bs[buf][nr][8 * g + 16];
            bfg[ns] = SHUF16(lo, hi);
        }
#pragma unroll
        for (int ms = 0; ms < 4; ++ms)
#pragma unroll
            for (int ns = 0; ns < 2; ++ns)
                acc[ms][ns] = WMMA_BF16(af[ms], bfg[ns], acc[ms][ns]);

        wait_async0();    // next tile fully in LDS (this wave's copies)
        __syncthreads();  // ... and everyone else's
    }

    // epilogue: C layout c[r] = C[r + 8*g][lane&15]
#pragma unroll
    for (int ms = 0; ms < 4; ++ms) {
#pragma unroll
        for (int ns = 0; ns < 2; ++ns) {
            const int row = mBase + waveM * 64 + ms * 16 + 8 * g;
            const int col = nBase + waveN * 32 + ns * 16 + ln16;
            float bv = 0.f;
            if constexpr (EPI >= 1) bv = bias[col];
#pragma unroll
            for (int r = 0; r < 8; ++r) {
                float v = acc[ms][ns][r];
                const size_t o = (size_t)(row + r) * N + col;
                if constexpr (EPI == 0) {
                    outB[o] = f2bf(v);
                } else if constexpr (EPI == 1) {
                    v += bv;
                    v = v > 0.f ? v : 0.f;
                    outB[o] = f2bf(v);
                } else {
                    outF[o] = v + bv + res[o];
                }
            }
        }
    }
}

// ---------------------------------------------------------------------------
// Causal flash attention. One wave per 16-query tile of one (b,h).
// Q,K: (B,S,D) bf16 head-interleaved; Vt: (B,H,HD,S) bf16; Ctx: (B,S,D) bf16.
// scores = Q*K^T/8 (two chained k=32 WMMAs), online softmax, P*V via WMMA
// with P bounced through per-wave LDS tile (C-layout -> A-layout).
// ---------------------------------------------------------------------------
__global__ __launch_bounds__(128) void attn_kernel(const bf16_t* __restrict__ Qm,
                                                   const bf16_t* __restrict__ Km,
                                                   const bf16_t* __restrict__ Vt,
                                                   bf16_t* __restrict__ Ctx) {
    __shared__ __align__(16) bf16_t P[4][16][40];
    const int tid  = threadIdx.x;
    const int lane = tid & 31;
    const int wid  = tid >> 5;
    const int g = lane >> 4, n = lane & 15;

    const int tile = blockIdx.x * 4 + wid;   // 0..8191
    const int qt = tile & 127;               // S/16 tiles
    const int bh = tile >> 7;                // 0..63
    const int h = bh & 15, b = bh >> 4;
    const int q0 = qt << 4;

    const bf16_t* qrow  = Qm + ((size_t)(b * Sc + q0 + n)) * Dc + h * HDc;
    const bf16_t* kbase = Km + ((size_t)b * Sc) * Dc + h * HDc;
    const bf16_t* vbase = Vt + ((size_t)bh * HDc) * Sc;

    // Q A-fragments for kk = 0, 32
    v16bf qa[2];
#pragma unroll
    for (int kk2 = 0; kk2 < 2; ++kk2) {
        bf16x8 lo = *(const bf16x8*)(qrow + kk2 * 32 + 8 * g);
        bf16x8 hi = *(const bf16x8*)(qrow + kk2 * 32 + 8 * g + 16);
        qa[kk2] = SHUF16(lo, hi);
    }

    float mrow[8], lrow[8];
    v8f acc[4];
#pragma unroll
    for (int r = 0; r < 8; ++r) { mrow[r] = -1e30f; lrow[r] = 0.f; }
#pragma unroll
    for (int t = 0; t < 4; ++t) acc[t] = zero_v8f();

    for (int j0 = 0; j0 <= q0; j0 += 32) {
        const bool have2 = (j0 + 16) <= (q0 + 15);

        v8f s0 = zero_v8f();
        {
            const bf16_t* kr = kbase + ((size_t)(j0 + n)) * Dc;
#pragma unroll
            for (int kk2 = 0; kk2 < 2; ++kk2) {
                bf16x8 lo = *(const bf16x8*)(kr + kk2 * 32 + 8 * g);
                bf16x8 hi = *(const bf16x8*)(kr + kk2 * 32 + 8 * g + 16);
                v16bf kf = SHUF16(lo, hi);
                s0 = WMMA_BF16(qa[kk2], kf, s0);
            }
        }
        v8f s1 = zero_v8f();
        if (have2) {
            const bf16_t* kr = kbase + ((size_t)(j0 + 16 + n)) * Dc;
#pragma unroll
            for (int kk2 = 0; kk2 < 2; ++kk2) {
                bf16x8 lo = *(const bf16x8*)(kr + kk2 * 32 + 8 * g);
                bf16x8 hi = *(const bf16x8*)(kr + kk2 * 32 + 8 * g + 16);
                v16bf kf = SHUF16(lo, hi);
                s1 = WMMA_BF16(qa[kk2], kf, s1);
            }
        }

        // online softmax over the 32-key block; rows striped r+8g, cols = lane&15
#pragma unroll
        for (int r = 0; r < 8; ++r) {
            const int qi = q0 + r + 8 * g;
            float sv0 = s0[r] * 0.125f;
            if (j0 + n > qi) sv0 = -1e30f;
            float sv1 = -1e30f;
            if (have2) {
                sv1 = s1[r] * 0.125f;
                if (j0 + 16 + n > qi) sv1 = -1e30f;
            }
            float rmax = fmaxf(sv0, sv1);
#pragma unroll
            for (int mm = 1; mm < 16; mm <<= 1) rmax = fmaxf(rmax, __shfl_xor(rmax, mm, 32));
            const float newm = fmaxf(mrow[r], rmax);
            const float p0 = __expf(sv0 - newm);
            const float p1 = have2 ? __expf(sv1 - newm) : 0.f;
            float rsum = p0 + p1;
#pragma unroll
            for (int mm = 1; mm < 16; mm <<= 1) rsum += __shfl_xor(rsum, mm, 32);
            const float sc = __expf(mrow[r] - newm);
            lrow[r] = lrow[r] * sc + rsum;
            mrow[r] = newm;
#pragma unroll
            for (int t = 0; t < 4; ++t) acc[t][r] *= sc;
            P[wid][r + 8 * g][n]      = f2bf(p0);
            P[wid][r + 8 * g][16 + n] = f2bf(p1);
        }

        // wait for LDS stores from this wave before re-reading in A-layout
        asm volatile("s_wait_dscnt 0" ::: "memory");

        bf16x8 plo = *(const bf16x8*)&P[wid][n][16 * g];
        bf16x8 phi = *(const bf16x8*)&P[wid][n][16 * g + 8];
        v16bf pf = SHUF16(plo, phi);

        int vcol = j0 + 16 * g;
        if (vcol > Sc - 16) vcol = Sc - 16; // masked half only; P is zero there
#pragma unroll
        for (int t = 0; t < 4; ++t) {
            const bf16_t* vr = vbase + ((size_t)(t * 16 + n)) * Sc + vcol;
            bf16x8 lo = *(const bf16x8*)vr;
            bf16x8 hi = *(const bf16x8*)(vr + 8);
            v16bf vf = SHUF16(lo, hi);
            acc[t] = WMMA_BF16(pf, vf, acc[t]);
        }
    }

    // normalize and store ctx (bf16, head-interleaved)
#pragma unroll
    for (int t = 0; t < 4; ++t) {
#pragma unroll
        for (int r = 0; r < 8; ++r) {
            const float val = acc[t][r] / lrow[r];
            Ctx[((size_t)(b * Sc + q0 + r + 8 * g)) * Dc + h * HDc + t * 16 + n] = f2bf(val);
        }
    }
}

// ---------------------------------------------------------------------------
// Launcher. Workspace layout (needs ~136 MiB, offsets 256B-aligned):
//   [0,16M)    lnb   bf16 LN output (reused for ln(x) and ln(h))
//   [16,48M)   hbuf  f32 residual stream h
//   [48,72M)   bf16 weights: wq,wk,wv,wu (2M each), w1 (8M), w2 (8M)
//   [72,136M)  qb(16) kb(16) vb(16) vtb(16); ctx aliases vb; ff1 overlays all
// ---------------------------------------------------------------------------
extern "C" void kernel_launch(void* const* d_in, const int* in_sizes, int n_in,
                              void* d_out, int out_size, void* d_ws, size_t ws_size,
                              hipStream_t stream) {
    (void)in_sizes; (void)n_in; (void)out_size; (void)ws_size;
    const float* x  = (const float*)d_in[0];
    const float* wq = (const float*)d_in[1];
    const float* wk = (const float*)d_in[2];
    const float* wv = (const float*)d_in[3];
    const float* wu = (const float*)d_in[4];
    const float* bu = (const float*)d_in[5];
    const float* w1 = (const float*)d_in[6];
    const float* b1 = (const float*)d_in[7];
    const float* w2 = (const float*)d_in[8];
    const float* b2 = (const float*)d_in[9];
    float* out = (float*)d_out;

    char* ws = (char*)d_ws;
    const size_t MiB = 1ull << 20;
    bf16_t* lnb  = (bf16_t*)(ws + 0 * MiB);
    float*  hbuf = (float*)(ws + 16 * MiB);
    bf16_t* wqb  = (bf16_t*)(ws + 48 * MiB);
    bf16_t* wkb  = (bf16_t*)(ws + 50 * MiB);
    bf16_t* wvb  = (bf16_t*)(ws + 52 * MiB);
    bf16_t* wub  = (bf16_t*)(ws + 54 * MiB);
    bf16_t* w1b  = (bf16_t*)(ws + 56 * MiB);
    bf16_t* w2b  = (bf16_t*)(ws + 64 * MiB);
    bf16_t* qb   = (bf16_t*)(ws + 72 * MiB);
    bf16_t* kb   = (bf16_t*)(ws + 88 * MiB);
    bf16_t* vb   = (bf16_t*)(ws + 104 * MiB);
    bf16_t* vtb  = (bf16_t*)(ws + 120 * MiB);
    bf16_t* ctxb = vb;                       // v dead after transpose
    bf16_t* ff1b = (bf16_t*)(ws + 72 * MiB); // q/k/v/vt dead after proj GEMM

    // 1) weights -> bf16
    f32_to_bf16_kernel<<<4096, 256, 0, stream>>>(wq, wqb, Dc * Dc);
    f32_to_bf16_kernel<<<4096, 256, 0, stream>>>(wk, wkb, Dc * Dc);
    f32_to_bf16_kernel<<<4096, 256, 0, stream>>>(wv, wvb, Dc * Dc);
    f32_to_bf16_kernel<<<4096, 256, 0, stream>>>(wu, wub, Dc * Dc);
    f32_to_bf16_kernel<<<16384, 256, 0, stream>>>(w1, w1b, Fc * Dc);
    f32_to_bf16_kernel<<<16384, 256, 0, stream>>>(w2, w2b, Dc * Fc);

    // 2) ln(x)
    ln_kernel<<<Mrows, 256, 0, stream>>>(x, lnb);

    // 3) q,k,v projections
    gemm_nt<0><<<dim3(8, 64), 256, 0, stream>>>(lnb, wqb, nullptr, nullptr, qb, nullptr,
                                                Mrows, Dc, Dc);
    gemm_nt<0><<<dim3(8, 64), 256, 0, stream>>>(lnb, wkb, nullptr, nullptr, kb, nullptr,
                                                Mrows, Dc, Dc);
    gemm_nt<0><<<dim3(8, 64), 256, 0, stream>>>(lnb, wvb, nullptr, nullptr, vb, nullptr,
                                                Mrows, Dc, Dc);

    // 4) V -> (B,H,HD,S)
    transpose_v_kernel<<<32768, 256, 0, stream>>>(vb, vtb);

    // 5) causal flash attention
    attn_kernel<<<2048, 128, 0, stream>>>(qb, kb, vtb, ctxb);

    // 6) h = x + ctx @ wu^T + bu   (f32)
    gemm_nt<2><<<dim3(8, 64), 256, 0, stream>>>(ctxb, wub, bu, x, nullptr, hbuf,
                                                Mrows, Dc, Dc);

    // 7) ln(h)
    ln_kernel<<<Mrows, 256, 0, stream>>>(hbuf, lnb);

    // 8) ff1 = relu(ln(h) @ w1^T + b1)  (bf16)
    gemm_nt<1><<<dim3(32, 64), 256, 0, stream>>>(lnb, w1b, b1, nullptr, ff1b, nullptr,
                                                 Mrows, Fc, Dc);

    // 9) out = h + ff1 @ w2^T + b2  (f32)
    gemm_nt<2><<<dim3(8, 64), 256, 0, stream>>>(ff1b, w2b, b2, hbuf, nullptr, out,
                                                Mrows, Dc, Fc);
}
